// CustomMultiHeadAttention_9818295238866
// MI455X (gfx1250) — matrix operand order repaired and checked by reading, not compiled, not run on hardware
//
#include <hip/hip_runtime.h>

typedef __attribute__((ext_vector_type(16))) _Float16 v16h;
typedef __attribute__((ext_vector_type(8)))  _Float16 v8h;
typedef __attribute__((ext_vector_type(4)))  _Float16 v4h;
typedef __attribute__((ext_vector_type(8)))  float    v8f;
typedef __attribute__((ext_vector_type(4)))  float    v4f;

#define DM   1024
#define SEQ  2048
#define NB   2
#define NH   16
#define DK   64
#define NROWS (NB*SEQ)   // 4096

__device__ __forceinline__ v8f vzero8() {
  v8f z = {0.f,0.f,0.f,0.f,0.f,0.f,0.f,0.f};
  return z;
}

__device__ __forceinline__ v8f wmma_f16(v16h a, v16h b, v8f c) {
  // D = A(16x32 f16) * B(32x16 f16) + C(16x16 f32)
  return __builtin_amdgcn_wmma_f32_16x16x32_f16(false, a, false, b, (short)0, c, false, false);
}

// A operand: per-lane halves [p, p+8) and [p+16, p+24)   (p = row + k + (lane>>4)*8)
__device__ __forceinline__ v16h load_a_f16(const _Float16* p) {
  union { v16h v; v8h h[2]; } u;
  u.h[0] = *(const v8h*)(p);
  u.h[1] = *(const v8h*)(p + 16);
  return u.v;
}
// B operand: per-lane halves [p, p+8) and [p+16, p+24)  (p = col_row + k + (lane>>4)*8)
__device__ __forceinline__ v16h load_b_f16(const _Float16* p) {
  union { v16h v; v8h h[2]; } u;
  u.h[0] = *(const v8h*)(p);
  u.h[1] = *(const v8h*)(p + 16);
  return u.v;
}

__global__ void cvt_f32_f16(const float* __restrict__ s, _Float16* __restrict__ d, int n) {
  int i = (blockIdx.x * blockDim.x + threadIdx.x) * 4;
  if (i < n) {
    v4f f = *(const v4f*)(s + i);
    v4h h;
    h[0] = (_Float16)f[0]; h[1] = (_Float16)f[1];
    h[2] = (_Float16)f[2]; h[3] = (_Float16)f[3];
    *(v4h*)(d + i) = h;
  }
}

// C = A[4096x1024] @ W^T[1024x1024] + bias.  One wave computes a 64x64 tile.
// MODE 0: A=f32, out f16 at [B,H,S,DK]       (Q, K projections)
// MODE 1: A=f32, out f16 at [B,H,DK,S]       (V projection, pre-transposed)
// MODE 2: A=f16, out f32 row-major [M,N]     (output projection)
template <int MODE>
__global__ void __launch_bounds__(32)
gemm_wmma_k(const void* __restrict__ Aptr,
            const _Float16* __restrict__ W,
            const float* __restrict__ bias,
            void* __restrict__ Out)
{
  const int lane = threadIdx.x;
  const int hf = lane >> 4;   // lane half (0/1)
  const int ln = lane & 15;
  const int M0 = blockIdx.y * 64;
  const int N0 = blockIdx.x * 64;

  v8f acc[4][4];
#pragma unroll
  for (int i = 0; i < 4; ++i)
#pragma unroll
    for (int j = 0; j < 4; ++j) acc[i][j] = vzero8();

  for (int k = 0; k < DM; k += 32) {
    v16h a[4];
#pragma unroll
    for (int mt = 0; mt < 4; ++mt) {
      const int row = M0 + mt * 16 + ln;
      if (MODE == 2) {
        const _Float16* A = (const _Float16*)Aptr;
        a[mt] = load_a_f16(A + (size_t)row * DM + k + hf * 8);
      } else {
        const float* A = (const float*)Aptr;
        const float* p = A + (size_t)row * DM + k + hf * 8;
        v4f f0 = *(const v4f*)p;
        v4f f1 = *(const v4f*)(p + 4);
        v4f f2 = *(const v4f*)(p + 16);
        v4f f3 = *(const v4f*)(p + 20);
        v16h t;
#pragma unroll
        for (int j = 0; j < 4; ++j) {
          t[j]      = (_Float16)f0[j];
          t[4 + j]  = (_Float16)f1[j];
          t[8 + j]  = (_Float16)f2[j];
          t[12 + j] = (_Float16)f3[j];
        }
        a[mt] = t;
      }
    }
#pragma unroll
    for (int nt = 0; nt < 4; ++nt) {
      // B[k][n] = W[n][k]: lane ln is column N0+nt*16+ln; k-halfs in A's table order
      v16h b = load_b_f16(W + (size_t)(N0 + nt * 16 + ln) * DM + k + hf * 8);
#pragma unroll
      for (int mt = 0; mt < 4; ++mt)
        acc[mt][nt] = wmma_f16(a[mt], b, acc[mt][nt]);
    }
  }

#pragma unroll
  for (int nt = 0; nt < 4; ++nt) {
    const int n = N0 + nt * 16 + ln;
    const float bn = bias[n];
#pragma unroll
    for (int mt = 0; mt < 4; ++mt) {
#pragma unroll
      for (int r = 0; r < 8; ++r) {
        const int m = M0 + mt * 16 + hf * 8 + r;   // C layout: M = r + 8*(lane>>4)
        const float val = acc[mt][nt][r] + bn;
        if (MODE == 2) {
          ((float*)Out)[(size_t)m * DM + n] = val;
        } else {
          const int bi = m >> 11, s = m & (SEQ - 1);
          const int h = n >> 6,  d = n & (DK - 1);
          const size_t idx = (MODE == 0)
              ? ((((size_t)bi * NH + h) * SEQ + s) * DK + d)
              : ((((size_t)bi * NH + h) * DK + d) * SEQ + s);
          ((_Float16*)Out)[idx] = (_Float16)val;
        }
      }
    }
  }
}

// One wave per (b, h, 16-query tile). Full-row softmax through 64KB LDS.
__global__ void __launch_bounds__(32)
attn_k(const _Float16* __restrict__ Q,   // [B,H,S,DK]
       const _Float16* __restrict__ K,   // [B,H,S,DK]
       const _Float16* __restrict__ Vt,  // [B,H,DK,S]
       _Float16* __restrict__ O)         // [B,S,DM]
{
  __shared__ _Float16 sP[16 * SEQ];      // 64 KB: scores -> exp(P) (unnormalized)
  const int lane = threadIdx.x;
  const int hf = lane >> 4;
  const int ln = lane & 15;
  const int q0 = blockIdx.x * 16;
  const int h  = blockIdx.y;
  const int b  = blockIdx.z;

  const _Float16* Qh = Q  + (((size_t)b * NH + h) * SEQ) * DK;
  const _Float16* Kh = K  + (((size_t)b * NH + h) * SEQ) * DK;
  const _Float16* Vh = Vt + (((size_t)b * NH + h) * DK) * SEQ;

  // Q tile A-operands for k=0..31 and k=32..63 (d_k = 64)
  const v16h aQ0 = load_a_f16(Qh + (size_t)(q0 + ln) * DK + hf * 8);
  const v16h aQ1 = load_a_f16(Qh + (size_t)(q0 + ln) * DK + 32 + hf * 8);

  // Phase 1: S = Q K^T * scale -> LDS (f16, row-major [m][n])
  for (int n0 = 0; n0 < SEQ; n0 += 16) {
    const _Float16* kp = Kh + (size_t)(n0 + ln) * DK + hf * 8;
    v16h bK0 = load_b_f16(kp);
    v16h bK1 = load_b_f16(kp + 32);
    v8f c = vzero8();
    c = wmma_f16(aQ0, bK0, c);
    c = wmma_f16(aQ1, bK1, c);
#pragma unroll
    for (int r = 0; r < 8; ++r)
      sP[(hf * 8 + r) * SEQ + n0 + ln] = (_Float16)(c[r] * 0.125f);
  }
  __syncthreads();

  // Phase 2: row softmax stats. Lane L handles row L>>1, half (L&1) of 2048.
  _Float16* rp = sP + (lane >> 1) * SEQ + (lane & 1) * 1024;
  float mx = -3.0e38f;
  for (int i = 0; i < 1024; i += 8) {
    v8h v = *(const v8h*)(rp + i);
#pragma unroll
    for (int j = 0; j < 8; ++j) mx = fmaxf(mx, (float)v[j]);
  }
  mx = fmaxf(mx, __shfl_xor(mx, 1, 32));
  float sum = 0.0f;
  for (int i = 0; i < 1024; i += 8) {
    v8h v = *(const v8h*)(rp + i);
    v8h e;
#pragma unroll
    for (int j = 0; j < 8; ++j) {
      float ev = __expf((float)v[j] - mx);
      sum += ev;
      e[j] = (_Float16)ev;
    }
    *(v8h*)(rp + i) = e;
  }
  sum += __shfl_xor(sum, 1, 32);
  const float inv = 1.0f / sum;
  __syncthreads();

  // Phase 3: O = exp(P) @ V   (A from LDS, B contiguous from V^T)
  v8f cO[4];
#pragma unroll
  for (int i = 0; i < 4; ++i) cO[i] = vzero8();
  for (int k = 0; k < SEQ; k += 32) {
    v16h aP = load_a_f16(sP + (size_t)ln * SEQ + k + hf * 8);
#pragma unroll
    for (int nt = 0; nt < 4; ++nt) {
      v16h bV = load_b_f16(Vh + (size_t)(nt * 16 + ln) * SEQ + k + hf * 8);
      cO[nt] = wmma_f16(aP, bV, cO[nt]);
    }
  }

  // normalize (1/sum lives in lanes 2m, 2m+1) and store [B,S,H*DK]
  float invr[8];
#pragma unroll
  for (int r = 0; r < 8; ++r) invr[r] = __shfl(inv, (hf * 8 + r) * 2, 32);
#pragma unroll
  for (int nt = 0; nt < 4; ++nt) {
#pragma unroll
    for (int r = 0; r < 8; ++r) {
      const int s = q0 + hf * 8 + r;
      O[((size_t)b * SEQ + s) * DM + h * DK + nt * 16 + ln] =
          (_Float16)(cO[nt][r] * invr[r]);
    }
  }
}

extern "C" void kernel_launch(void* const* d_in, const int* in_sizes, int n_in,
                              void* d_out, int out_size, void* d_ws, size_t ws_size,
                              hipStream_t stream) {
  (void)in_sizes; (void)n_in; (void)out_size; (void)ws_size;
  const float* q  = (const float*)d_in[0];
  const float* ky = (const float*)d_in[1];
  const float* vl = (const float*)d_in[2];
  const float* Wq = (const float*)d_in[3];
  const float* bq = (const float*)d_in[4];
  const float* Wk = (const float*)d_in[5];
  const float* bk = (const float*)d_in[6];
  const float* Wv = (const float*)d_in[7];
  const float* bv = (const float*)d_in[8];
  const float* Wo = (const float*)d_in[9];
  const float* bo = (const float*)d_in[10];

  _Float16* ws = (_Float16*)d_ws;
  const size_t MW = (size_t)DM * DM;          // 1M halfs per weight
  const size_t MT = (size_t)NB * NH * SEQ * DK; // 4M halfs per tensor
  _Float16* Wq16 = ws + 0 * MW;
  _Float16* Wk16 = ws + 1 * MW;
  _Float16* Wv16 = ws + 2 * MW;
  _Float16* Wo16 = ws + 3 * MW;
  _Float16* Q16  = ws + 4 * MW;
  _Float16* K16  = Q16 + MT;
  _Float16* Vt16 = K16 + MT;
  _Float16* O16  = Vt16 + MT;   // total ws use: 40 MB

  const int ct = 256;
  const int cb = (int)(MW / 4 / ct);
  cvt_f32_f16<<<cb, ct, 0, stream>>>(Wq, Wq16, (int)MW);
  cvt_f32_f16<<<cb, ct, 0, stream>>>(Wk, Wk16, (int)MW);
  cvt_f32_f16<<<cb, ct, 0, stream>>>(Wv, Wv16, (int)MW);
  cvt_f32_f16<<<cb, ct, 0, stream>>>(Wo, Wo16, (int)MW);

  dim3 gg(DM / 64, NROWS / 64);  // (16, 64): 64x64 tile per wave
  gemm_wmma_k<0><<<gg, 32, 0, stream>>>(q,  Wq16, bq, Q16);
  gemm_wmma_k<0><<<gg, 32, 0, stream>>>(ky, Wk16, bk, K16);
  gemm_wmma_k<1><<<gg, 32, 0, stream>>>(vl, Wv16, bv, Vt16);

  attn_k<<<dim3(SEQ / 16, NH, NB), 32, 0, stream>>>(Q16, K16, Vt16, O16);

  gemm_wmma_k<2><<<gg, 32, 0, stream>>>(O16, Wo16, bo, d_out);
}
